// GroupedQueryAttention_26860725469381
// MI455X (gfx1250) — matrix operand order repaired and checked by reading, not compiled, not run on hardware
//
#include <hip/hip_runtime.h>
#include <hip/hip_bf16.h>

typedef __attribute__((ext_vector_type(16))) __bf16 v16bf;
typedef __attribute__((ext_vector_type(8)))  float  v8f;
typedef __attribute__((ext_vector_type(4)))  int    v4i;
typedef unsigned short u16;

#define DIM     2048
#define HEADS   16
#define KVHEADS 4
#define HD      128
#define SEQ     2048
#define BATCH   2
#define KVDIM   (KVHEADS * HD)   // 512
#define MROWS   (BATCH * SEQ)    // 4096

// ---------- CDNA5 async global->LDS path (guarded; falls back to sync copy) ----------
#if defined(__AMDGCN__) && __has_builtin(__builtin_amdgcn_global_load_async_to_lds_b128) && \
    __has_builtin(__builtin_amdgcn_s_wait_asynccnt)
#define HAVE_ASYNC_LDS 1
#else
#define HAVE_ASYNC_LDS 0
#endif

typedef __attribute__((address_space(1))) v4i* gv4i_p;
typedef __attribute__((address_space(3))) v4i* lv4i_p;

__device__ __forceinline__ void async_cp_b128(const u16* g, u16* l) {
#if HAVE_ASYNC_LDS
    __builtin_amdgcn_global_load_async_to_lds_b128((gv4i_p)g, (lv4i_p)l, 0, 0);
#else
    *reinterpret_cast<uint4*>(l) = *reinterpret_cast<const uint4*>(g);
#endif
}

template<int N>
__device__ __forceinline__ void async_wait() {
#if HAVE_ASYNC_LDS
    __builtin_amdgcn_s_wait_asynccnt(N);
#endif
}

// ---------- helpers ----------

__device__ __forceinline__ u16 f2bf(float f) {
    unsigned int x = __float_as_uint(f);
    unsigned int r = x + 0x7fffu + ((x >> 16) & 1u);   // round-to-nearest-even
    return (u16)(r >> 16);
}

// Build a 16x32 (A) or 32x16 (B) bf16 WMMA fragment from two 16-byte chunks.
__device__ __forceinline__ v16bf ld_frag(const u16* p0, const u16* p1) {
    union { v16bf v; uint4 u[2]; } r;
    r.u[0] = *reinterpret_cast<const uint4*>(p0);
    r.u[1] = *reinterpret_cast<const uint4*>(p1);
    return r.v;
}

__device__ __forceinline__ v8f wmma_bf16(v16bf a, v16bf b, v8f c) {
    return __builtin_amdgcn_wmma_f32_16x16x32_bf16(false, a, false, b, (short)0, c,
                                                   false, false);
}

// ---------- conversion kernels ----------

__global__ void cvt_f32_bf16(const float* __restrict__ in, u16* __restrict__ out, int n4) {
    int i = blockIdx.x * blockDim.x + threadIdx.x;
    if (i < n4) {
        float4 f = reinterpret_cast<const float4*>(in)[i];
        ushort4 o;
        o.x = f2bf(f.x); o.y = f2bf(f.y); o.z = f2bf(f.z); o.w = f2bf(f.w);
        reinterpret_cast<ushort4*>(out)[i] = o;
    }
}

// Wt[n*K + k] = bf16(W[k*N + n]); K x N input, N x K output.
__global__ void transpose_f32_bf16(const float* __restrict__ W, u16* __restrict__ Wt,
                                   int K, int N) {
    __shared__ u16 tile[32][33];
    int n0 = blockIdx.x * 32, k0 = blockIdx.y * 32;
    #pragma unroll
    for (int i = threadIdx.y; i < 32; i += 8)
        tile[i][threadIdx.x] = f2bf(W[(size_t)(k0 + i) * N + n0 + threadIdx.x]);
    __syncthreads();
    #pragma unroll
    for (int i = threadIdx.y; i < 32; i += 8)
        Wt[(size_t)(n0 + i) * K + k0 + threadIdx.x] = tile[threadIdx.x][i];
}

// ---------- GEMM: C[M x Nout] = A[M x K](bf16) * Wt[Nout x K](bf16, pre-transposed) ----
// 128x128 block tile, K-step 32, 8 waves (4 over M x 2 over N), 2x4 WMMA tiles / wave.
// Double-buffered LDS with async global->LDS staging overlapped with WMMA.

template<bool F32OUT>
__global__ __launch_bounds__(256)
void gemm_bf16(const u16* __restrict__ A, const u16* __restrict__ Wt,
               void* __restrict__ Cout, int M, int Nout, int K) {
    __shared__ u16 As[2][128][40];   // padded stride: 80B rows
    __shared__ u16 Bs[2][128][40];

    const int tid  = threadIdx.x;
    const int lane = tid & 31;
    const int wave = tid >> 5;
    const int wm   = wave & 3;        // 4 waves over M (32 rows each)
    const int wn   = wave >> 2;       // 2 waves over N (64 cols each)
    const int lm   = lane & 15;
    const int hf   = lane >> 4;
    const int mBase = blockIdx.y * 128;
    const int nBase = blockIdx.x * 128;

    v8f acc[2][4];
    #pragma unroll
    for (int mi = 0; mi < 2; ++mi)
        #pragma unroll
        for (int ni = 0; ni < 4; ++ni)
            #pragma unroll
            for (int i = 0; i < 8; ++i) acc[mi][ni][i] = 0.0f;

    const int ldRow = tid >> 1;         // 0..127
    const int ldCol = (tid & 1) * 16;   // 0 / 16

    auto stage = [&](int buf, int kb) {
        const u16* ga = A  + (size_t)(mBase + ldRow) * K + kb + ldCol;
        const u16* gb = Wt + (size_t)(nBase + ldRow) * K + kb + ldCol;
        async_cp_b128(ga,     &As[buf][ldRow][ldCol]);
        async_cp_b128(ga + 8, &As[buf][ldRow][ldCol + 8]);
        async_cp_b128(gb,     &Bs[buf][ldRow][ldCol]);
        async_cp_b128(gb + 8, &Bs[buf][ldRow][ldCol + 8]);
    };

    const int nk = K >> 5;
    stage(0, 0);                                 // prologue: tile 0 in flight

    for (int ki = 0; ki < nk; ++ki) {
        const int cur = ki & 1;
        __syncthreads();                         // buffer 1-cur free for overwrite
        if (ki + 1 < nk) {
            stage(1 - cur, (ki + 1) << 5);
            async_wait<4>();                     // oldest 4 copies (buf cur) retired
        } else {
            async_wait<0>();
        }
        __syncthreads();                         // tile cur visible to all waves

        v16bf af[2], bfr[4];
        #pragma unroll
        for (int mi = 0; mi < 2; ++mi) {
            int r = wm * 32 + mi * 16 + lm;
            af[mi] = ld_frag(&As[cur][r][8 * hf], &As[cur][r][16 + 8 * hf]);
        }
        #pragma unroll
        for (int ni = 0; ni < 4; ++ni) {
            int r = wn * 64 + ni * 16 + lm;
            const u16* p = &Bs[cur][r][8 * hf];
            bfr[ni] = ld_frag(p, p + 16);
        }
        #pragma unroll
        for (int mi = 0; mi < 2; ++mi)
            #pragma unroll
            for (int ni = 0; ni < 4; ++ni)
                acc[mi][ni] = wmma_bf16(af[mi], bfr[ni], acc[mi][ni]);
    }

    #pragma unroll
    for (int mi = 0; mi < 2; ++mi)
        #pragma unroll
        for (int ni = 0; ni < 4; ++ni) {
            int col = nBase + wn * 64 + ni * 16 + lm;
            #pragma unroll
            for (int i = 0; i < 8; ++i) {
                int row = mBase + wm * 32 + mi * 16 + i + 8 * hf;
                float v = acc[mi][ni][i];
                if (F32OUT)
                    reinterpret_cast<float*>(Cout)[(size_t)row * Nout + col] = v;
                else
                    reinterpret_cast<u16*>(Cout)[(size_t)row * Nout + col] = f2bf(v);
            }
        }
}

// ---------- GQA flash attention ----------
// grid = (SEQ/128, HEADS, BATCH); block = 256 (8 waves, 16 query rows each).

__global__ __launch_bounds__(256)
void gqa_attention(const u16* __restrict__ qg, const u16* __restrict__ kg,
                   const u16* __restrict__ vg, u16* __restrict__ og) {
    __shared__ u16 Ks[64][136];      // K tile  [key][d]
    __shared__ u16 Vt[128][80];      // V tile transposed [d][key]
    __shared__ u16 Pt[8][16][72];    // per-wave P scratch [qrow][key]

    const int tid  = threadIdx.x;
    const int lane = tid & 31;
    const int wave = tid >> 5;
    const int lm   = lane & 15;
    const int hf   = lane >> 4;
    const int qt = blockIdx.x, h = blockIdx.y, b = blockIdx.z;
    const int hk = h >> 2;                     // GROUP = 4
    const int qBase = qt * 128 + wave * 16;
    const float scale = 0.08838834764831845f;  // 1/sqrt(128)

    // Q fragments (A layout), 4 chunks of 32 over d = 128, loaded once.
    v16bf qa[4];
    {
        const u16* qp = qg + (size_t)(b * SEQ + qBase + lm) * DIM + h * HD;
        #pragma unroll
        for (int dc = 0; dc < 4; ++dc) {
            const u16* p0 = qp + dc * 32 + 8 * hf;
            qa[dc] = ld_frag(p0, p0 + 16);
        }
    }

    v8f oacc[8];
    #pragma unroll
    for (int nt = 0; nt < 8; ++nt)
        #pragma unroll
        for (int i = 0; i < 8; ++i) oacc[nt][i] = 0.0f;
    float mrow[8], lrow[8];
    #pragma unroll
    for (int i = 0; i < 8; ++i) { mrow[i] = -3.0e38f; lrow[i] = 0.0f; }

    const int ldr = tid >> 2;          // 0..63 : key row
    const int ldc = (tid & 3) * 32;    // d chunk

    for (int kb = 0; kb < SEQ; kb += 64) {
        // stage V into regs (needs in-register transpose)
        const u16* kp = kg + (size_t)(b * SEQ + kb + ldr) * KVDIM + hk * HD + ldc;
        const u16* vp = vg + (size_t)(b * SEQ + kb + ldr) * KVDIM + hk * HD + ldc;
        union { uint4 u[4]; u16 s[32]; } vv;
        #pragma unroll
        for (int ii = 0; ii < 4; ++ii)
            vv.u[ii] = *reinterpret_cast<const uint4*>(vp + 8 * ii);
        __syncthreads();                         // previous tile readers done
        #pragma unroll
        for (int ii = 0; ii < 4; ++ii)           // K tile: async global->LDS
            async_cp_b128(kp + 8 * ii, &Ks[ldr][ldc + 8 * ii]);
        #pragma unroll
        for (int i = 0; i < 32; ++i)             // transpose V into [d][key]
            Vt[ldc + i][ldr] = vv.s[i];
        async_wait<0>();
        __syncthreads();

        // S = Q * K^T  (4 key sub-tiles of 16)
        v8f sacc[4];
        #pragma unroll
        for (int j = 0; j < 4; ++j) {
            #pragma unroll
            for (int i = 0; i < 8; ++i) sacc[j][i] = 0.0f;
            #pragma unroll
            for (int dc = 0; dc < 4; ++dc) {
                const u16* p = &Ks[16 * j + lm][32 * dc + 8 * hf];
                sacc[j] = wmma_bf16(qa[dc], ld_frag(p, p + 16), sacc[j]);
            }
        }

        // online softmax on C layout: row = i + 8*hf, col = lm
        float mx[8];
        #pragma unroll
        for (int i = 0; i < 8; ++i) {
            float m = -3.0e38f;
            #pragma unroll
            for (int j = 0; j < 4; ++j) {
                sacc[j][i] *= scale;
                m = fmaxf(m, sacc[j][i]);
            }
            #pragma unroll
            for (int msk = 1; msk < 16; msk <<= 1)
                m = fmaxf(m, __shfl_xor(m, msk, 32));
            mx[i] = m;
        }
        float al[8];
        #pragma unroll
        for (int i = 0; i < 8; ++i) {
            float mn = fmaxf(mrow[i], mx[i]);
            al[i] = __expf(mrow[i] - mn);
            mrow[i] = mn;
        }
        float rs[8];
        #pragma unroll
        for (int i = 0; i < 8; ++i) rs[i] = 0.0f;
        #pragma unroll
        for (int j = 0; j < 4; ++j)
            #pragma unroll
            for (int i = 0; i < 8; ++i) {
                float p = __expf(sacc[j][i] - mrow[i]);
                sacc[j][i] = p;
                rs[i] += p;
            }
        #pragma unroll
        for (int i = 0; i < 8; ++i) {
            float r = rs[i];
            #pragma unroll
            for (int msk = 1; msk < 16; msk <<= 1)
                r += __shfl_xor(r, msk, 32);
            lrow[i] = lrow[i] * al[i] + r;
        }
        #pragma unroll
        for (int nt = 0; nt < 8; ++nt)
            #pragma unroll
            for (int i = 0; i < 8; ++i) oacc[nt][i] *= al[i];

        // P: C layout -> A layout via per-wave LDS round-trip
        #pragma unroll
        for (int j = 0; j < 4; ++j)
            #pragma unroll
            for (int i = 0; i < 8; ++i)
                Pt[wave][i + 8 * hf][16 * j + lm] = f2bf(sacc[j][i]);
        __threadfence_block();
        v16bf pa[2];
        #pragma unroll
        for (int kc = 0; kc < 2; ++kc) {
            const u16* p0 = &Pt[wave][lm][32 * kc + 8 * hf];
            pa[kc] = ld_frag(p0, p0 + 16);
        }

        // O += P * V  (B frags from transposed V tile: lane = d, K contiguous)
        #pragma unroll
        for (int kc = 0; kc < 2; ++kc)
            #pragma unroll
            for (int nt = 0; nt < 8; ++nt) {
                const u16* p = &Vt[16 * nt + lm][32 * kc + 8 * hf];
                oacc[nt] = wmma_bf16(pa[kc], ld_frag(p, p + 16), oacc[nt]);
            }
    }

    // epilogue: normalize and store [b*SEQ + n][h*HD + d] bf16
    #pragma unroll
    for (int i = 0; i < 8; ++i) {
        float inv = 1.0f / lrow[i];
        size_t row = (size_t)(b * SEQ + qBase + i + 8 * hf);
        #pragma unroll
        for (int nt = 0; nt < 8; ++nt)
            og[row * DIM + h * HD + nt * 16 + lm] = f2bf(oacc[nt][i] * inv);
    }
}

// ---------- host launch ----------

extern "C" void kernel_launch(void* const* d_in, const int* in_sizes, int n_in,
                              void* d_out, int out_size, void* d_ws, size_t ws_size,
                              hipStream_t stream) {
    const float* x  = (const float*)d_in[0];
    const float* Wq = (const float*)d_in[1];
    const float* Wk = (const float*)d_in[2];
    const float* Wv = (const float*)d_in[3];
    const float* Wo = (const float*)d_in[4];
    float* out = (float*)d_out;

    char* ws = (char*)d_ws;
    size_t off = 0;
    auto carve = [&](size_t bytes) {
        void* p = ws + off;
        off += (bytes + 255) & ~(size_t)255;
        return p;
    };
    u16* xb  = (u16*)carve((size_t)MROWS * DIM * 2);    // x bf16
    u16* wqt = (u16*)carve((size_t)DIM * DIM * 2);      // Wq^T
    u16* wkt = (u16*)carve((size_t)KVDIM * DIM * 2);    // Wk^T
    u16* wvt = (u16*)carve((size_t)KVDIM * DIM * 2);    // Wv^T
    u16* wot = (u16*)carve((size_t)DIM * DIM * 2);      // Wo^T
    u16* qb  = (u16*)carve((size_t)MROWS * DIM * 2);
    u16* kb  = (u16*)carve((size_t)MROWS * KVDIM * 2);
    u16* vb  = (u16*)carve((size_t)MROWS * KVDIM * 2);
    u16* ob  = (u16*)carve((size_t)MROWS * DIM * 2);

    // 1. convert x, transpose+convert weights
    {
        int n4 = MROWS * DIM / 4;
        cvt_f32_bf16<<<(n4 + 255) / 256, 256, 0, stream>>>(x, xb, n4);
        dim3 blk(32, 8);
        transpose_f32_bf16<<<dim3(DIM / 32,   DIM / 32), blk, 0, stream>>>(Wq, wqt, DIM, DIM);
        transpose_f32_bf16<<<dim3(KVDIM / 32, DIM / 32), blk, 0, stream>>>(Wk, wkt, DIM, KVDIM);
        transpose_f32_bf16<<<dim3(KVDIM / 32, DIM / 32), blk, 0, stream>>>(Wv, wvt, DIM, KVDIM);
        transpose_f32_bf16<<<dim3(DIM / 32,   DIM / 32), blk, 0, stream>>>(Wo, wot, DIM, DIM);
    }

    // 2. Q/K/V projections (bf16 out)
    gemm_bf16<false><<<dim3(DIM / 128,   MROWS / 128), 256, 0, stream>>>(xb, wqt, qb, MROWS, DIM,   DIM);
    gemm_bf16<false><<<dim3(KVDIM / 128, MROWS / 128), 256, 0, stream>>>(xb, wkt, kb, MROWS, KVDIM, DIM);
    gemm_bf16<false><<<dim3(KVDIM / 128, MROWS / 128), 256, 0, stream>>>(xb, wvt, vb, MROWS, KVDIM, DIM);

    // 3. grouped-query flash attention
    gqa_attention<<<dim3(SEQ / 128, HEADS, BATCH), 256, 0, stream>>>(qb, kb, vb, ob);

    // 4. output projection (f32 out)
    gemm_bf16<true><<<dim3(DIM / 128, MROWS / 128), 256, 0, stream>>>(ob, wot, out, MROWS, DIM, DIM);
}
